// GCNLayer_75857712382545
// MI455X (gfx1250) — matrix operand order repaired and checked by reading, compiled, not run on hardware
//
#include <hip/hip_runtime.h>

// CDNA5 / gfx1250 GCN layer:
//   msgs = norm * (xs @ W1^T + b1 + (xs*xd) @ W2^T + b2)
//   h    = scatter_sum(msgs, edges_dst);  out = LeakyReLU(h, 0.2)
// Matrix work runs on v_wmma_f32_16x16x32_bf16 (f32 accumulate).
// Persistent blocks: weights staged into LDS B-fragment layout once per block.

typedef __attribute__((ext_vector_type(16))) __bf16 v16bf;
typedef __attribute__((ext_vector_type(2)))  __bf16 v2bf;
typedef __attribute__((ext_vector_type(2)))  float  v2f;
typedef __attribute__((ext_vector_type(8)))  float  v8f;

#define WAVES_PER_BLOCK 8
#define TILE_E 16
#define MAX_BLOCKS 2048

union BFrag {
  v16bf v;
  unsigned short s[16];
};

// f32x2 -> bf16x2: vector fptrunc so the backend can select a packed convert
// (v_cvt_pk_bf16_f32-class) when gfx1250 provides one.
__device__ __forceinline__ void cvt_pk_bf16(unsigned short* dst, float lo, float hi) {
  v2f  f = {lo, hi};
  v2bf b = __builtin_convertvector(f, v2bf);
  __builtin_memcpy(dst, &b, 4);
}

__global__ __launch_bounds__(256) void gcn_zero(float* __restrict__ p, int n4) {
  int i = blockIdx.x * blockDim.x + threadIdx.x;
  if (i < n4) ((float4*)p)[i] = make_float4(0.f, 0.f, 0.f, 0.f);
}

__global__ __launch_bounds__(256) void gcn_leaky(float* __restrict__ p, int n4) {
  int i = blockIdx.x * blockDim.x + threadIdx.x;
  if (i < n4) {
    float4 v = ((const float4*)p)[i];
    v.x = v.x > 0.f ? v.x : 0.2f * v.x;
    v.y = v.y > 0.f ? v.y : 0.2f * v.y;
    v.z = v.z > 0.f ? v.z : 0.2f * v.z;
    v.w = v.w > 0.f ? v.w : 0.2f * v.w;
    ((float4*)p)[i] = v;
  }
}

// One wave (32 lanes) processes a TILE_E=16 edge tile per loop iteration.
// B fragments (W1^T, W2^T in bf16) live in LDS, pre-swizzled once per block:
//   set = mat*8 + ntile*2 + kstep ;  bfrag[set][lane][e]
//   B element e of lane L  ->  K = kstep*32 + (L>=16 ? 16 : 0) + e,  N = ntile*16 + (L&15)
__global__ __launch_bounds__(256) void gcn_edge_wmma(
    const float* __restrict__ srcEmb, const float* __restrict__ dstEmb,
    const float* __restrict__ norm,
    const float* __restrict__ W1, const float* __restrict__ b1,
    const float* __restrict__ W2, const float* __restrict__ b2,
    const int* __restrict__ eSrc, const int* __restrict__ eDst,
    float* __restrict__ h, int nSrc, int E, int numTiles)
{
  __shared__ __align__(32) unsigned short bfrag[16][32][16];   // 16 KB
  __shared__ float bsum[64];

  // --- stage W1^T / W2^T into LDS in B-fragment layout (bf16), pairs of e ---
  for (int i = threadIdx.x; i < 16 * 32 * 8; i += 256) {
    int e2   = (i & 7) * 2;
    int lane = (i >> 3) & 31;
    int set  = i >> 8;
    int mat  = set >> 3;
    int rem  = set & 7;
    int t    = rem >> 1;          // ntile
    int s    = rem & 1;           // kstep
    int k    = s * 32 + ((lane & 16) ? 16 : 0) + e2;
    int n    = t * 16 + (lane & 15);
    const float* W = mat ? W2 : W1;
    // B[k][n] = W[n][k]
    cvt_pk_bf16(&bfrag[set][lane][e2], W[n * 64 + k], W[n * 64 + k + 1]);
  }
  if (threadIdx.x < 64) bsum[threadIdx.x] = b1[threadIdx.x] + b2[threadIdx.x];
  __syncthreads();

  int lane = threadIdx.x & 31;
  int wave = threadIdx.x >> 5;
  int row  = lane & 15;                          // edge row within tile
  int hi   = (lane >> 4) & 1;                    // lane>=16 half
  int stride = gridDim.x * WAVES_PER_BLOCK;

  for (int tile = blockIdx.x * WAVES_PER_BLOCK + wave; tile < numTiles; tile += stride) {
    int tileBase = tile * TILE_E;

    // --- gather this lane's edge row (clamped; masked by norm=0 on tail) ---
    int edge = tileBase + row;
    if (edge >= E) edge = E - 1;
    int sn = eSrc[edge];
    int dn = eDst[edge];
    const float* xsrow = (sn < nSrc) ? (srcEmb + (long)sn * 64)
                                     : (dstEmb + (long)(sn - nSrc) * 64);
    const float* xdrow = (dn < nSrc) ? (srcEmb + (long)dn * 64)
                                     : (dstEmb + (long)(dn - nSrc) * 64);

    // --- build A fragments: A1 = bf16(xs), A2 = bf16(xs*xd) ---
    // element e of lane L -> feature kstep*32 + (L>=16?16:0) + e
    BFrag a1[2], a2[2];
#pragma unroll
    for (int s = 0; s < 2; ++s) {
      int base = s * 32 + hi * 16;
      float xsv[16], xdv[16];
      *(float4*)&xsv[0]  = *(const float4*)(xsrow + base);
      *(float4*)&xsv[4]  = *(const float4*)(xsrow + base + 4);
      *(float4*)&xsv[8]  = *(const float4*)(xsrow + base + 8);
      *(float4*)&xsv[12] = *(const float4*)(xsrow + base + 12);
      *(float4*)&xdv[0]  = *(const float4*)(xdrow + base);
      *(float4*)&xdv[4]  = *(const float4*)(xdrow + base + 4);
      *(float4*)&xdv[8]  = *(const float4*)(xdrow + base + 8);
      *(float4*)&xdv[12] = *(const float4*)(xdrow + base + 12);
#pragma unroll
      for (int e = 0; e < 16; e += 2) {
        cvt_pk_bf16(&a1[s].s[e], xsv[e], xsv[e + 1]);
        cvt_pk_bf16(&a2[s].s[e], xsv[e] * xdv[e], xsv[e + 1] * xdv[e + 1]);
      }
    }

    // --- per-edge scalars for the epilogue (32-bit scatter offsets) ---
    float    nrm[8];
    unsigned doff[8];                      // dnode*64, fits in 32 bits
#pragma unroll
    for (int r = 0; r < 8; ++r) {
      int idx = tileBase + hi * 8 + r;
      if (idx < E) { nrm[r] = norm[idx]; doff[r] = (unsigned)eDst[idx] * 64u; }
      else         { nrm[r] = 0.f;       doff[r] = 0u; }
    }

    // --- 4 output tiles of 16x16; 4 WMMAs each (2 kstep x 2 matrices) ---
#pragma unroll
    for (int t = 0; t < 4; ++t) {
      v8f acc = {0.f, 0.f, 0.f, 0.f, 0.f, 0.f, 0.f, 0.f};
      const v16bf* B1a = (const v16bf*)&bfrag[0 + t * 2 + 0][lane][0];
      const v16bf* B1b = (const v16bf*)&bfrag[0 + t * 2 + 1][lane][0];
      const v16bf* B2a = (const v16bf*)&bfrag[8 + t * 2 + 0][lane][0];
      const v16bf* B2b = (const v16bf*)&bfrag[8 + t * 2 + 1][lane][0];
      acc = __builtin_amdgcn_wmma_f32_16x16x32_bf16(false, a1[0].v, false, *B1a,
                                                    (short)0, acc, false, false);
      acc = __builtin_amdgcn_wmma_f32_16x16x32_bf16(false, a1[1].v, false, *B1b,
                                                    (short)0, acc, false, false);
      acc = __builtin_amdgcn_wmma_f32_16x16x32_bf16(false, a2[0].v, false, *B2a,
                                                    (short)0, acc, false, false);
      acc = __builtin_amdgcn_wmma_f32_16x16x32_bf16(false, a2[1].v, false, *B2b,
                                                    (short)0, acc, false, false);

      // D layout: lane holds column N=t*16+(lane&15); VGPR r -> M = r + hi*8
      int feat = t * 16 + row;
      float bs = bsum[feat];
#pragma unroll
      for (int r = 0; r < 8; ++r) {
        float val = (acc[r] + bs) * nrm[r];
        // 32-bit element offset from uniform base -> saddr-form global atomic
        atomicAdd(h + (doff[r] + (unsigned)feat), val);
      }
    }
  }
}

extern "C" void kernel_launch(void* const* d_in, const int* in_sizes, int n_in,
                              void* d_out, int out_size, void* d_ws, size_t ws_size,
                              hipStream_t stream) {
  const float* srcEmb = (const float*)d_in[0];
  const float* dstEmb = (const float*)d_in[1];
  const float* norm   = (const float*)d_in[2];
  const float* W1     = (const float*)d_in[3];
  const float* b1     = (const float*)d_in[4];
  const float* W2     = (const float*)d_in[5];
  const float* b2     = (const float*)d_in[6];
  const int*   eSrc   = (const int*)d_in[7];
  const int*   eDst   = (const int*)d_in[8];

  int nSrc = in_sizes[0] / 64;
  int E    = in_sizes[2];
  float* out = (float*)d_out;

  int n4 = out_size / 4;
  gcn_zero<<<(n4 + 255) / 256, 256, 0, stream>>>(out, n4);

  int numTiles = (E + TILE_E - 1) / TILE_E;
  int blocks   = (numTiles + WAVES_PER_BLOCK - 1) / WAVES_PER_BLOCK;
  if (blocks > MAX_BLOCKS) blocks = MAX_BLOCKS;   // persistent grid
  gcn_edge_wmma<<<blocks, 256, 0, stream>>>(srcEmb, dstEmb, norm, W1, b1, W2, b2,
                                            eSrc, eDst, out, nSrc, E, numTiles);

  gcn_leaky<<<(n4 + 255) / 256, 256, 0, stream>>>(out, n4);
}
